// SelfAttention_82197084111079
// MI455X (gfx1250) — matrix operand order repaired and checked by reading, not compiled, not run on hardware
//
#include <hip/hip_runtime.h>

#define DM 1024
#define NH 16
#define HD 64
#define BB 4
#define SS 2048
#define ROWS (BB * SS)          // 8192
#define KT32 (DM / 32)          // 32 k-tiles of depth 32

#if defined(__has_builtin)
#if __has_builtin(__builtin_amdgcn_global_load_async_to_lds_b128)
#define USE_ASYNC_LDS 1
#endif
#endif
#ifndef USE_ASYNC_LDS
#define USE_ASYNC_LDS 0
#endif

typedef __attribute__((ext_vector_type(16))) __bf16 v16bf;
typedef __attribute__((ext_vector_type(8)))  float  v8f;
typedef int v4i_vs __attribute__((vector_size(16)));
#define AS_GLOBAL __attribute__((address_space(1)))
#define AS_LDS    __attribute__((address_space(3)))

union Frag {
    v16bf v;
    uint4 q[2];
    unsigned short us[16];
};

__device__ __forceinline__ unsigned short f32_to_bf16(float f) {
    union { float f; unsigned int u; } c; c.f = f;
    unsigned int u = c.u;
    unsigned int r = u + 0x7FFFu + ((u >> 16) & 1u);   // round-to-nearest-even
    return (unsigned short)(r >> 16);
}

__device__ __forceinline__ v8f wmma_bf16(v16bf a, v16bf b, v8f c) {
    return __builtin_amdgcn_wmma_f32_16x16x32_bf16(false, a, false, b, (short)0, c, false, false);
}

__device__ __forceinline__ v8f zero8() {
    v8f z = {0.f, 0.f, 0.f, 0.f, 0.f, 0.f, 0.f, 0.f};
    return z;
}

// ---------------- 1) f32 -> bf16 convert (x) ----------------
__global__ __launch_bounds__(256) void cvt_bf16_kernel(const float4* __restrict__ x,
                                                       uint2* __restrict__ out, int n4) {
    int i = blockIdx.x * blockDim.x + threadIdx.x;
    if (i >= n4) return;
    float4 a = x[i];
    uint2 r;
    r.x = (unsigned)f32_to_bf16(a.x) | ((unsigned)f32_to_bf16(a.y) << 16);
    r.y = (unsigned)f32_to_bf16(a.z) | ((unsigned)f32_to_bf16(a.w) << 16);
    out[i] = r;
}

// ---------------- 2) weight pack: f32 [K,N] -> bf16 B-fragments ----------------
// Fragment f = nt*kTiles + kt holds the 32x16 tile (K=kt*32.., N=nt*16..) in
// B-fragment order: lane l -> column N = nt*16 + (l&15), K values kt*32 + h8 + 0..7
// then kt*32 + 16 + h8 + 0..7 (h8 = (l>=16)?8:0). 512 bf16 per fragment, lane-contiguous.
__global__ __launch_bounds__(256) void pack_w_kernel(const float* __restrict__ W,
                                                     unsigned short* __restrict__ P,
                                                     int N, int kTiles, int nFrags) {
    int wid = (blockIdx.x * blockDim.x + threadIdx.x) >> 5;
    if (wid >= nFrags) return;
    int lane = threadIdx.x & 31;
    int nt = wid / kTiles, kt = wid % kTiles;
    int col = lane & 15;
    int half16 = (lane >> 4) << 4;
    Frag f;
#pragma unroll
    for (int i = 0; i < 16; ++i) {
        int kk = kt * 32 + (half16 >> 1) + (i & 7) + ((i >> 3) << 4);
        f.us[i] = f32_to_bf16(W[(size_t)kk * N + (nt * 16 + col)]);
    }
    uint4* dst = reinterpret_cast<uint4*>(P + (size_t)wid * 512 + lane * 16);
    dst[0] = f.q[0];
    dst[1] = f.q[1];
}

// ---------------- 3) QKV GEMM: 32x64 tile per wave, double-buffered k-loop ----------------
__global__ __launch_bounds__(256) void qkv_gemm_kernel(const unsigned short* __restrict__ xbf,
                                                       const unsigned short* __restrict__ wp,
                                                       unsigned short* __restrict__ qb,
                                                       unsigned short* __restrict__ kbuf,
                                                       unsigned short* __restrict__ vtb) {
    int wid = (blockIdx.x * blockDim.x + threadIdx.x) >> 5;
    if (wid >= 256 * 48) return;
    int lane = threadIdx.x & 31;
    int mt = wid & 255;
    int ng = wid >> 8;                 // 0..47  (64 output cols each)
    int m0 = mt << 5, n0 = ng << 6;
    int col = lane & 15;
    int h8 = (lane >> 4) << 3;
    const unsigned short* arow0 = xbf + (size_t)(m0 + col) * DM;
    const unsigned short* arow1 = arow0 + (size_t)16 * DM;

    Frag a0[2], a1[2], bf[2][4];
    auto loadA = [&](int slot, int kt) {
        const unsigned short* ap0 = arow0 + kt * 32 + h8;
        a0[slot].q[0] = *reinterpret_cast<const uint4*>(ap0);
        a0[slot].q[1] = *reinterpret_cast<const uint4*>(ap0 + 16);
        const unsigned short* ap1 = arow1 + kt * 32 + h8;
        a1[slot].q[0] = *reinterpret_cast<const uint4*>(ap1);
        a1[slot].q[1] = *reinterpret_cast<const uint4*>(ap1 + 16);
    };
    auto loadB = [&](int slot, int kt) {
        const unsigned short* bp = wp + ((size_t)ng * 4 * KT32 + kt) * 512 + lane * 16;
#pragma unroll
        for (int n = 0; n < 4; ++n) {
            bf[slot][n].q[0] = *reinterpret_cast<const uint4*>(bp);
            bf[slot][n].q[1] = *reinterpret_cast<const uint4*>(bp + 8);
            bp += (size_t)KT32 * 512;
        }
    };

    v8f acc0[4] = {zero8(), zero8(), zero8(), zero8()};
    v8f acc1[4] = {zero8(), zero8(), zero8(), zero8()};
    loadA(0, 0);
    loadB(0, 0);
    for (int kt = 0; kt < KT32; ++kt) {
        int cur = kt & 1, nxt = cur ^ 1;
        if (kt + 1 < KT32) {           // issue next-iteration loads before consuming
            loadA(nxt, kt + 1);
            loadB(nxt, kt + 1);
        }
#pragma unroll
        for (int n = 0; n < 4; ++n) {
            acc0[n] = wmma_bf16(a0[cur].v, bf[cur][n].v, acc0[n]);
            acc1[n] = wmma_bf16(a1[cur].v, bf[cur][n].v, acc1[n]);
        }
    }
    // epilogue: n0 block of 64 cols lies entirely in one of Q/K/V and one head
    int b   = m0 / SS;
    int s0  = m0 % SS;
    int sec = n0 / DM;                 // 0=Q 1=K 2=V
    int h   = (n0 % DM) / HD;
    size_t bh = (size_t)b * NH + h;
#pragma unroll
    for (int r = 0; r < 2; ++r) {
        v8f* acc = r ? acc1 : acc0;
        int sbase = s0 + r * 16;
#pragma unroll
        for (int n = 0; n < 4; ++n) {
            int d = n * 16 + col;
#pragma unroll
            for (int j = 0; j < 8; ++j) {
                int srow = sbase + j + h8;
                unsigned short bv = f32_to_bf16(acc[n][j]);
                if (sec == 0)      qb  [(bh * SS + srow) * HD + d] = bv;
                else if (sec == 1) kbuf[(bh * SS + srow) * HD + d] = bv;
                else               vtb [(bh * HD + d) * SS + srow] = bv;   // V transposed
            }
        }
    }
}

// ---------------- 4) Flash attention: one wave = one (b,h, 16-query tile) ----------------
__global__ __launch_bounds__(256) void attn_kernel(const unsigned short* __restrict__ qb,
                                                   const unsigned short* __restrict__ kbuf,
                                                   const unsigned short* __restrict__ vtb,
                                                   unsigned short* __restrict__ ob) {
    __shared__ __align__(16) unsigned short lds[8][16][32];
#if USE_ASYNC_LDS
    __shared__ __align__(16) unsigned short vstage[8][4][32][16];   // 32 KB
#endif
    int wid = (blockIdx.x * blockDim.x + threadIdx.x) >> 5;
    if (wid >= 64 * (SS / 16)) return;
    int w = (threadIdx.x >> 5) & 7;
    int lane = threadIdx.x & 31;
    int bh = wid >> 7;                 // 0..63
    int qt = wid & 127;
    int q0 = qt << 4;
    int col = lane & 15;
    int h8 = (lane >> 4) << 3;
    int half16 = (lane >> 4) << 4;

    const unsigned short* Q  = qb   + (size_t)bh * SS * HD;
    const unsigned short* Km = kbuf + (size_t)bh * SS * HD;
    const unsigned short* Vt = vtb  + (size_t)bh * HD * SS;

    // Q A-fragments for d=0..31 and d=32..63
    Frag qf0, qf1;
    {
        const unsigned short* qp = Q + (size_t)(q0 + col) * HD;
        qf0.q[0] = *reinterpret_cast<const uint4*>(qp + h8);
        qf0.q[1] = *reinterpret_cast<const uint4*>(qp + h8 + 16);
        qf1.q[0] = *reinterpret_cast<const uint4*>(qp + 32 + h8);
        qf1.q[1] = *reinterpret_cast<const uint4*>(qp + 48 + h8);
    }

    float m[8], l[8];
    v8f of[4] = {zero8(), zero8(), zero8(), zero8()};
#pragma unroll
    for (int j = 0; j < 8; ++j) { m[j] = -1e30f; l[j] = 0.f; }

    int nc = (q0 + 16 + 31) >> 5;      // causal: key chunks of 32
    for (int kc = 0; kc < nc; ++kc) {
        int k0 = kc << 5;
#if USE_ASYNC_LDS
        // stream this chunk's V strips into LDS via the async engine; they are
        // consumed only after the QK^T WMMAs + softmax below (overlap).
#pragma unroll
        for (int n = 0; n < 4; ++n) {
            const unsigned short* vp = Vt + (size_t)(n * 16 + col) * SS + k0 + half16;
            AS_GLOBAL v4i_vs* gsrc = (AS_GLOBAL v4i_vs*)vp;
            AS_LDS    v4i_vs* ldst = (AS_LDS v4i_vs*)&vstage[w][n][lane][0];
            __builtin_amdgcn_global_load_async_to_lds_b128(gsrc,     ldst,     0, 0);
            __builtin_amdgcn_global_load_async_to_lds_b128(gsrc + 1, ldst + 1, 0, 0);
        }
#endif
        __builtin_prefetch(Km + (size_t)(k0 + 32 + col) * HD, 0, 1);
        v8f sc[2];
#pragma unroll
        for (int u = 0; u < 2; ++u) {
            int kbase = k0 + u * 16;
            const unsigned short* kp = Km + (size_t)(kbase + col) * HD + h8;
            Frag kf0, kf1;
            kf0.q[0] = *reinterpret_cast<const uint4*>(kp);
            kf0.q[1] = *reinterpret_cast<const uint4*>(kp + 16);
            kf1.q[0] = *reinterpret_cast<const uint4*>(kp + 32);
            kf1.q[1] = *reinterpret_cast<const uint4*>(kp + 48);
            v8f s = zero8();
            s = wmma_bf16(qf0.v, kf0.v, s);
            s = wmma_bf16(qf1.v, kf1.v, s);
            sc[u] = s;
        }
        // scale + causal mask
        bool diag = (k0 + 31) > q0;
#pragma unroll
        for (int u = 0; u < 2; ++u)
#pragma unroll
            for (int j = 0; j < 8; ++j) {
                float v = sc[u][j] * 0.125f;          // 1/sqrt(64)
                if (diag && (k0 + u * 16 + col) > (q0 + j + h8)) v = -1e30f;
                sc[u][j] = v;
            }
        // online softmax row stats (rows j+h8 live in this lane-half; reduce over 16 lanes)
        float mnew[8], rs[8], alpha[8];
#pragma unroll
        for (int j = 0; j < 8; ++j) {
            float v = fmaxf(sc[0][j], sc[1][j]);
#pragma unroll
            for (int off = 1; off < 16; off <<= 1) v = fmaxf(v, __shfl_xor(v, off, 16));
            mnew[j] = fmaxf(m[j], v);
        }
#pragma unroll
        for (int j = 0; j < 8; ++j) {
            float p0 = __expf(sc[0][j] - mnew[j]);
            float p1 = __expf(sc[1][j] - mnew[j]);
            lds[w][j + h8][col]      = f32_to_bf16(p0);
            lds[w][j + h8][16 + col] = f32_to_bf16(p1);
            float rv = p0 + p1;
#pragma unroll
            for (int off = 1; off < 16; off <<= 1) rv += __shfl_xor(rv, off, 16);
            rs[j] = rv;
        }
#pragma unroll
        for (int j = 0; j < 8; ++j) {
            alpha[j] = __expf(m[j] - mnew[j]);
            l[j] = l[j] * alpha[j] + rs[j];
            m[j] = mnew[j];
        }
#pragma unroll
        for (int n = 0; n < 4; ++n)
#pragma unroll
            for (int j = 0; j < 8; ++j) of[n][j] *= alpha[j];
        // P (16x32) back as A-fragment via LDS
        asm volatile("s_wait_dscnt 0" ::: "memory");
        Frag pf;
        pf.q[0] = *reinterpret_cast<const uint4*>(&lds[w][col][half16]);
        pf.q[1] = *reinterpret_cast<const uint4*>(&lds[w][col][half16 + 8]);
        // P @ V over 4 head-dim column tiles
#if USE_ASYNC_LDS
        asm volatile("s_wait_asynccnt 0" ::: "memory");
#pragma unroll
        for (int n = 0; n < 4; ++n) {
            Frag vf;
            vf.q[0] = *reinterpret_cast<const uint4*>(&vstage[w][n][lane][0]);
            vf.q[1] = *reinterpret_cast<const uint4*>(&vstage[w][n][lane][8]);
            of[n] = wmma_bf16(pf.v, vf.v, of[n]);
        }
#else
#pragma unroll
        for (int n = 0; n < 4; ++n) {
            const unsigned short* vp = Vt + (size_t)(n * 16 + col) * SS + k0 + half16;
            Frag vf;
            vf.q[0] = *reinterpret_cast<const uint4*>(vp);
            vf.q[1] = *reinterpret_cast<const uint4*>(vp + 8);
            of[n] = wmma_bf16(pf.v, vf.v, of[n]);
        }
#endif
    }
    // epilogue: normalize and store O as [b, s, h*64+d] bf16
    int b = bh >> 4, h = bh & 15;
#pragma unroll
    for (int n = 0; n < 4; ++n)
#pragma unroll
        for (int j = 0; j < 8; ++j) {
            int srow = q0 + j + h8;
            float v = of[n][j] / l[j];
            ob[((size_t)b * SS + srow) * DM + h * HD + n * 16 + col] = f32_to_bf16(v);
        }
}

// ---------------- 5) Output projection: 32x64 tile per wave, double-buffered ----------------
__global__ __launch_bounds__(256) void proj_gemm_kernel(const unsigned short* __restrict__ ob,
                                                        const unsigned short* __restrict__ wp,
                                                        float* __restrict__ out) {
    int wid = (blockIdx.x * blockDim.x + threadIdx.x) >> 5;
    if (wid >= 256 * 16) return;
    int lane = threadIdx.x & 31;
    int mt = wid & 255;
    int ng = wid >> 8;                 // 0..15
    int m0 = mt << 5, n0 = ng << 6;
    int col = lane & 15;
    int h8 = (lane >> 4) << 3;
    const unsigned short* arow0 = ob + (size_t)(m0 + col) * DM;
    const unsigned short* arow1 = arow0 + (size_t)16 * DM;

    Frag a0[2], a1[2], bf[2][4];
    auto loadA = [&](int slot, int kt) {
        const unsigned short* ap0 = arow0 + kt * 32 + h8;
        a0[slot].q[0] = *reinterpret_cast<const uint4*>(ap0);
        a0[slot].q[1] = *reinterpret_cast<const uint4*>(ap0 + 16);
        const unsigned short* ap1 = arow1 + kt * 32 + h8;
        a1[slot].q[0] = *reinterpret_cast<const uint4*>(ap1);
        a1[slot].q[1] = *reinterpret_cast<const uint4*>(ap1 + 16);
    };
    auto loadB = [&](int slot, int kt) {
        const unsigned short* bp = wp + ((size_t)ng * 4 * KT32 + kt) * 512 + lane * 16;
#pragma unroll
        for (int n = 0; n < 4; ++n) {
            bf[slot][n].q[0] = *reinterpret_cast<const uint4*>(bp);
            bf[slot][n].q[1] = *reinterpret_cast<const uint4*>(bp + 8);
            bp += (size_t)KT32 * 512;
        }
    };

    v8f acc0[4] = {zero8(), zero8(), zero8(), zero8()};
    v8f acc1[4] = {zero8(), zero8(), zero8(), zero8()};
    loadA(0, 0);
    loadB(0, 0);
    for (int kt = 0; kt < KT32; ++kt) {
        int cur = kt & 1, nxt = cur ^ 1;
        if (kt + 1 < KT32) {
            loadA(nxt, kt + 1);
            loadB(nxt, kt + 1);
        }
#pragma unroll
        for (int n = 0; n < 4; ++n) {
            acc0[n] = wmma_bf16(a0[cur].v, bf[cur][n].v, acc0[n]);
            acc1[n] = wmma_bf16(a1[cur].v, bf[cur][n].v, acc1[n]);
        }
    }
#pragma unroll
    for (int r = 0; r < 2; ++r) {
        v8f* acc = r ? acc1 : acc0;
        int mbase = m0 + r * 16;
#pragma unroll
        for (int n = 0; n < 4; ++n)
#pragma unroll
            for (int j = 0; j < 8; ++j)
                out[(size_t)(mbase + j + h8) * DM + n0 + n * 16 + col] = acc[n][j];
    }
}

// ---------------- launch ----------------
extern "C" void kernel_launch(void* const* d_in, const int* in_sizes, int n_in,
                              void* d_out, int out_size, void* d_ws, size_t ws_size,
                              hipStream_t stream) {
    const float* x     = (const float*)d_in[0];
    const float* w_qkv = (const float*)d_in[1];
    const float* w_out = (const float*)d_in[2];
    float* out = (float*)d_out;
    char* ws = (char*)d_ws;

    unsigned short* xbf = (unsigned short*)(ws + 0);            // 16 MB
    unsigned short* wqp = (unsigned short*)(ws + 16777216);     //  6 MB packed w_qkv
    unsigned short* wop = (unsigned short*)(ws + 23068672);     //  2 MB packed w_out
    unsigned short* qb  = (unsigned short*)(ws + 25165824);     // 16 MB Q  [bh,s,64]
    unsigned short* kb  = (unsigned short*)(ws + 41943040);     // 16 MB K  [bh,s,64]
    unsigned short* vtb = (unsigned short*)(ws + 58720256);     // 16 MB V^T[bh,64,s]
    unsigned short* ob  = (unsigned short*)(ws + 75497472);     // 16 MB attn out [b,s,d]

    // 1) x -> bf16
    int n4 = (ROWS * DM) / 4;                                   // 2,097,152
    cvt_bf16_kernel<<<n4 / 256, 256, 0, stream>>>((const float4*)x, (uint2*)xbf, n4);

    // 2) pack weights into B-fragment layout
    int fq = (3 * DM / 16) * KT32;                              // 6144 fragments
    pack_w_kernel<<<(fq * 32) / 256, 256, 0, stream>>>(w_qkv, wqp, 3 * DM, KT32, fq);
    int fo = (DM / 16) * KT32;                                  // 2048 fragments
    pack_w_kernel<<<(fo * 32) / 256, 256, 0, stream>>>(w_out, wop, DM, KT32, fo);

    // 3) QKV GEMM (256 m-tiles x 48 col-groups = 12288 waves, 32x64 each)
    qkv_gemm_kernel<<<(256 * 48 * 32) / 256, 256, 0, stream>>>(xbf, wqp, qb, kb, vtb);

    // 4) flash attention (64 bh x 128 q-tiles = 8192 waves)
    attn_kernel<<<(8192 * 32) / 256, 256, 0, stream>>>(qb, kb, vtb, ob);

    // 5) output projection (256 x 16 = 4096 waves, 32x64 each)
    proj_gemm_kernel<<<(256 * 16 * 32) / 256, 256, 0, stream>>>(ob, wop, out);
}
